// MultiHeadAttention_67731634258682
// MI455X (gfx1250) — matrix operand order repaired and checked by reading, not compiled, not run on hardware
//
#include <hip/hip_runtime.h>

typedef __attribute__((ext_vector_type(16))) _Float16 v16h;
typedef __attribute__((ext_vector_type(8)))  _Float16 v8h;
typedef __attribute__((ext_vector_type(2)))  _Float16 v2h;
typedef __attribute__((ext_vector_type(8)))  float    v8f;

#define DEV __device__ __forceinline__

// __builtin_amdgcn_cvt_pkrtz returns __fp16 ext_vector(2); bit-cast to v2h.
DEV v2h cvt_pk_h2(float x, float y) {
  return __builtin_bit_cast(v2h, __builtin_amdgcn_cvt_pkrtz(x, y));
}

DEV v8f wmma_f32_f16(v16h a, v16h b, v8f c) {
  // D = A(16x32 f16) * B(32x16 f16) + C(16x16 f32)
  return __builtin_amdgcn_wmma_f32_16x16x32_f16(
      /*neg_a=*/false, a, /*neg_b=*/false, b,
      /*c_mod=*/(short)0, c, /*reuse_a=*/false, /*reuse_b=*/false);
}

// ---- DPP16 16-lane butterfly reductions (no LDS, no dscnt waits) ----------
template <int CTRL>
DEV float dpp_mov_f(float x) {
  return __int_as_float(__builtin_amdgcn_update_dpp(
      0, __float_as_int(x), CTRL, 0xF, 0xF, true));
}
DEV float red16_max(float x) {
  x = fmaxf(x, dpp_mov_f<0xB1>(x));   // quad_perm [1,0,3,2]
  x = fmaxf(x, dpp_mov_f<0x4E>(x));   // quad_perm [2,3,0,1]
  x = fmaxf(x, dpp_mov_f<0x141>(x));  // row_half_mirror
  x = fmaxf(x, dpp_mov_f<0x140>(x));  // row_mirror
  return x;
}
DEV float red16_sum(float x) {
  x += dpp_mov_f<0xB1>(x);
  x += dpp_mov_f<0x4E>(x);
  x += dpp_mov_f<0x141>(x);
  x += dpp_mov_f<0x140>(x);
  return x;
}

// A fragment 16x32 f16 from row-major src (stride in halves, multiple of 8).
// ISA layout: lane (m = l&15, hi = l>>4); element 8g+e <-> k = 16g + 8hi + e.
DEV v16h load_frag_a(const _Float16* src, int stride) {
  const int lane = threadIdx.x & 31;
  const int m = lane & 15, hi = lane >> 4;
  const _Float16* row = src + m * stride;
  v8h lo = *reinterpret_cast<const v8h*>(row + 8 * hi);
  v8h hg = *reinterpret_cast<const v8h*>(row + 16 + 8 * hi);
  v16h r;
#pragma unroll
  for (int i = 0; i < 8; ++i) { r[i] = lo[i]; r[i + 8] = hg[i]; }
  return r;
}

// B fragment 32x16 f16 from N-major ("transposed") srcT[n][k] (stride in
// halves, multiple of 8). ISA layout: lane (n = l&15); element 8g+e <-> k = 16g + 8hi + e.
DEV v16h load_frag_b(const _Float16* srcT, int stride) {
  const int lane = threadIdx.x & 31;
  const int n = lane & 15, hi = lane >> 4;
  const _Float16* p = srcT + n * stride + 8 * hi;
  v8h lo = *reinterpret_cast<const v8h*>(p);
  v8h hg = *reinterpret_cast<const v8h*>(p + 16);
  v16h r;
#pragma unroll
  for (int i = 0; i < 8; ++i) { r[i] = lo[i]; r[i + 8] = hg[i]; }
  return r;
}

// ---------------------------------------------------------------------------
// GEMM: Out[M,N] = X[M,K] @ W[K,N] + bias, optionally scaled; X f32 or f16,
// Out f32 or f16. Block = 256 threads (8 waves), tile 128x128, k-chunk 32.
// ---------------------------------------------------------------------------
template <bool X_F16, bool OUT_F16>
__global__ __launch_bounds__(256, 2)
void gemm128(const void* __restrict__ Xv, const float* __restrict__ W,
             const float* __restrict__ bias, void* __restrict__ Outv,
             int M, int N, int K, float out_scale) {
  constexpr int LDA = 40;  // 32 + 8 pad halves (80B rows, 16B aligned)
  constexpr int LDB = 40;
  __shared__ _Float16 As[128 * LDA];  // [m][k]
  __shared__ _Float16 Bt[128 * LDB];  // [n][k] (transposed weight tile)

  const int tid = threadIdx.x;
  const int wave = tid >> 5, lane = tid & 31;
  const int ln = lane & 15, hi = lane >> 4;
  const int m0 = blockIdx.y * 128, n0 = blockIdx.x * 128;
  const int wm = (wave & 3) * 32, wn = (wave >> 2) * 64;

  v8f acc[2][4] = {};

  const int arow = tid >> 1;        // 128 rows, 2 threads per row
  const int acb  = (tid & 1) * 16;  // 16 halves each
  const int bk2  = (tid >> 4) * 2;  // even k pair (16 pairs of k)
  const int bnb  = (tid & 15) * 8;  // 8 columns each

  for (int k0 = 0; k0 < K; k0 += 32) {
    __syncthreads();
    // ---- stage A tile (convert to f16, packed stores) ----
    if constexpr (X_F16) {
      const _Float16* Xr = reinterpret_cast<const _Float16*>(Xv) +
                           (size_t)(m0 + arow) * K + k0 + acb;
      *reinterpret_cast<v8h*>(&As[arow * LDA + acb]) =
          *reinterpret_cast<const v8h*>(Xr);
      *reinterpret_cast<v8h*>(&As[arow * LDA + acb + 8]) =
          *reinterpret_cast<const v8h*>(Xr + 8);
      if (k0 + 32 < K) __builtin_prefetch(Xr + 32, 0, 1);
    } else {
      const float* Xr = reinterpret_cast<const float*>(Xv) +
                        (size_t)(m0 + arow) * K + k0 + acb;
      const float4* X4 = reinterpret_cast<const float4*>(Xr);
#pragma unroll
      for (int j = 0; j < 2; ++j) {
        float4 f0 = X4[2 * j], f1 = X4[2 * j + 1];
        v2h p0 = cvt_pk_h2(f0.x, f0.y);
        v2h p1 = cvt_pk_h2(f0.z, f0.w);
        v2h p2 = cvt_pk_h2(f1.x, f1.y);
        v2h p3 = cvt_pk_h2(f1.z, f1.w);
        v8h o;
        o[0] = p0[0]; o[1] = p0[1]; o[2] = p1[0]; o[3] = p1[1];
        o[4] = p2[0]; o[5] = p2[1]; o[6] = p3[0]; o[7] = p3[1];
        *reinterpret_cast<v8h*>(&As[arow * LDA + acb + j * 8]) = o;
      }
      if (k0 + 32 < K) __builtin_prefetch(Xr + 32, 0, 1);
    }
    // ---- stage B tile transposed: Bt[n][k] = W[k][n]; pack k-pairs -> b32 --
    {
      const float* W0 = W + (size_t)(k0 + bk2) * N + n0 + bnb;
      const float* W1 = W0 + N;
      const float4* W04 = reinterpret_cast<const float4*>(W0);
      const float4* W14 = reinterpret_cast<const float4*>(W1);
      float4 a0 = W04[0], a1 = W04[1];
      float4 c0 = W14[0], c1 = W14[1];
      *reinterpret_cast<v2h*>(&Bt[(bnb + 0) * LDB + bk2]) = cvt_pk_h2(a0.x, c0.x);
      *reinterpret_cast<v2h*>(&Bt[(bnb + 1) * LDB + bk2]) = cvt_pk_h2(a0.y, c0.y);
      *reinterpret_cast<v2h*>(&Bt[(bnb + 2) * LDB + bk2]) = cvt_pk_h2(a0.z, c0.z);
      *reinterpret_cast<v2h*>(&Bt[(bnb + 3) * LDB + bk2]) = cvt_pk_h2(a0.w, c0.w);
      *reinterpret_cast<v2h*>(&Bt[(bnb + 4) * LDB + bk2]) = cvt_pk_h2(a1.x, c1.x);
      *reinterpret_cast<v2h*>(&Bt[(bnb + 5) * LDB + bk2]) = cvt_pk_h2(a1.y, c1.y);
      *reinterpret_cast<v2h*>(&Bt[(bnb + 6) * LDB + bk2]) = cvt_pk_h2(a1.z, c1.z);
      *reinterpret_cast<v2h*>(&Bt[(bnb + 7) * LDB + bk2]) = cvt_pk_h2(a1.w, c1.w);
      if (k0 + 32 < K) __builtin_prefetch(W0 + 32 * (size_t)N, 0, 1);
    }
    __syncthreads();

    v16h af[2];
#pragma unroll
    for (int im = 0; im < 2; ++im)
      af[im] = load_frag_a(&As[(wm + 16 * im) * LDA], LDA);
#pragma unroll
    for (int in = 0; in < 4; ++in) {
      v16h bf = load_frag_b(&Bt[(wn + 16 * in) * LDB], LDB);
#pragma unroll
      for (int im = 0; im < 2; ++im)
        acc[im][in] = wmma_f32_f16(af[im], bf, acc[im][in]);
    }
  }

  // ---- epilogue: bias + scale, store ----
#pragma unroll
  for (int im = 0; im < 2; ++im) {
#pragma unroll
    for (int in = 0; in < 4; ++in) {
      const int col = n0 + wn + 16 * in + ln;
      const float bv = bias[col];
#pragma unroll
      for (int r = 0; r < 8; ++r) {
        const int row = m0 + wm + 16 * im + r + 8 * hi;
        const float val = (acc[im][in][r] + bv) * out_scale;
        if constexpr (OUT_F16)
          reinterpret_cast<_Float16*>(Outv)[(size_t)row * N + col] = (_Float16)val;
        else
          reinterpret_cast<float*>(Outv)[(size_t)row * N + col] = val;
      }
    }
  }
}

// ---------------------------------------------------------------------------
// Flash attention. Grid (S/128, H, B), block 256 (8 waves). One wave owns
// 16 q-rows; workgroup stages K (row-major) and V (transposed) tiles in LDS
// per 64-wide kv step. Q is pre-scaled by 1/sqrt(depth).
// ---------------------------------------------------------------------------
constexpr int S_LEN = 8192, DMODEL = 1024, DHEAD = 64;

__global__ __launch_bounds__(256, 2)
void flash_attn(const _Float16* __restrict__ Qh, const _Float16* __restrict__ Kh,
                const _Float16* __restrict__ Vh, _Float16* __restrict__ AO) {
  constexpr int KS  = 72;  // K tile row stride (64 + 8 pad halves)
  constexpr int VTS = 72;  // Vt row stride
  constexpr int PS  = 72;  // P scratch row stride
  __shared__ _Float16 Ks[DHEAD * KS];    // [kv][d] (row-major, 64 kv rows)
  __shared__ _Float16 Vt[DHEAD * VTS];   // [d][kv] (transposed)
  __shared__ _Float16 Pls[8 * 16 * PS];  // per-wave 16x64 P scratch

  const int tid = threadIdx.x;
  const int wave = tid >> 5, lane = tid & 31;
  const int ln = lane & 15, hi = lane >> 4;
  const int b = blockIdx.z, h = blockIdx.y;
  const int q0 = blockIdx.x * 128 + wave * 16;

  const _Float16* Qb = Qh + ((size_t)b * S_LEN + q0) * DMODEL + h * DHEAD;
  const _Float16* Kb = Kh + (size_t)b * S_LEN * DMODEL + h * DHEAD;
  const _Float16* Vb = Vh + (size_t)b * S_LEN * DMODEL + h * DHEAD;

  const v16h qa0 = load_frag_a(Qb, DMODEL);       // d 0..31
  const v16h qa1 = load_frag_a(Qb + 32, DMODEL);  // d 32..63

  v8f o[4] = {};
  float mrun[8], lrun[8];
#pragma unroll
  for (int r = 0; r < 8; ++r) { mrun[r] = -3.0e38f; lrun[r] = 0.f; }

  const int krow = tid & 63;         // K staging: kv row
  const int kdc  = (tid >> 6) * 16;  //            16-halves d chunk
  const int vkv2 = (tid & 31) * 2;   // V staging: even kv pair
  const int vdc  = (tid >> 5) * 8;   //            8 d values

  for (int s0 = 0; s0 < S_LEN; s0 += 64) {
    __syncthreads();
    // stage K tile row-major (straight copy, B-frag reads are contiguous)
    {
      const _Float16* kr = Kb + (size_t)(s0 + krow) * DMODEL + kdc;
      *reinterpret_cast<v8h*>(&Ks[krow * KS + kdc]) =
          *reinterpret_cast<const v8h*>(kr);
      *reinterpret_cast<v8h*>(&Ks[krow * KS + kdc + 8]) =
          *reinterpret_cast<const v8h*>(kr + 8);
    }
    // stage V tile transposed: Vt[d][kv], packed kv-pairs -> b32 stores
    {
      const _Float16* vr0 = Vb + (size_t)(s0 + vkv2) * DMODEL + vdc;
      const _Float16* vr1 = vr0 + DMODEL;
      v8h r0 = *reinterpret_cast<const v8h*>(vr0);
      v8h r1 = *reinterpret_cast<const v8h*>(vr1);
#pragma unroll
      for (int j = 0; j < 8; ++j) {
        v2h p; p[0] = r0[j]; p[1] = r1[j];
        *reinterpret_cast<v2h*>(&Vt[(vdc + j) * VTS + vkv2]) = p;
      }
    }
    if (s0 + 64 < S_LEN) {
      __builtin_prefetch(Kb + (size_t)(s0 + 64 + krow) * DMODEL + kdc, 0, 1);
      __builtin_prefetch(Vb + (size_t)(s0 + 64 + vkv2) * DMODEL + vdc, 0, 1);
    }
    __syncthreads();

    // S = Q K^T  (B-fragments from the shared K tile)
    v8f st[4];
#pragma unroll
    for (int t = 0; t < 4; ++t) {
      const _Float16* Kt = &Ks[16 * t * KS];
      v8f c = {};
      c = wmma_f32_f16(qa0, load_frag_b(Kt, KS), c);
      c = wmma_f32_f16(qa1, load_frag_b(Kt + 32, KS), c);
      st[t] = c;
    }

    // online softmax; row m = r + 8*hi lives in VGPR r across a 16-lane half
    float alpha[8];
#pragma unroll
    for (int r = 0; r < 8; ++r) {
      float mx = fmaxf(fmaxf(st[0][r], st[1][r]), fmaxf(st[2][r], st[3][r]));
      mx = red16_max(mx);
      const float mnew = fmaxf(mrun[r], mx);
      alpha[r] = __expf(mrun[r] - mnew);
      mrun[r] = mnew;
      float rs = 0.f;
#pragma unroll
      for (int t = 0; t < 4; ++t) {
        const float p = __expf(st[t][r] - mnew);
        st[t][r] = p; rs += p;
      }
      rs = red16_sum(rs);
      lrun[r] = lrun[r] * alpha[r] + rs;
    }
#pragma unroll
    for (int t = 0; t < 4; ++t)
#pragma unroll
      for (int r = 0; r < 8; ++r) o[t][r] *= alpha[r];

    // C-layout -> A-layout for P via per-wave LDS scratch
    _Float16* P = &Pls[wave * 16 * PS];
#pragma unroll
    for (int t = 0; t < 4; ++t)
#pragma unroll
      for (int r = 0; r < 8; ++r)
        P[(r + 8 * hi) * PS + 16 * t + ln] = (_Float16)st[t][r];
    asm volatile("s_wait_dscnt 0" ::: "memory");  // same-wave LDS RAW
    const v16h pa0 = load_frag_a(P, PS);        // kv 0..31
    const v16h pa1 = load_frag_a(P + 32, PS);   // kv 32..63

    // O += P V   (B-fragments from transposed V tile)
#pragma unroll
    for (int t = 0; t < 4; ++t) {
      o[t] = wmma_f32_f16(pa0, load_frag_b(&Vt[16 * t * VTS], VTS), o[t]);
      o[t] = wmma_f32_f16(pa1, load_frag_b(&Vt[16 * t * VTS] + 32, VTS), o[t]);
    }
  }

  // normalize and store attention output (f16, [B,S, h*64+d] layout)
  float inv[8];
#pragma unroll
  for (int r = 0; r < 8; ++r) inv[r] = 1.0f / lrun[r];
#pragma unroll
  for (int t = 0; t < 4; ++t)
#pragma unroll
    for (int r = 0; r < 8; ++r) {
      const size_t row = (size_t)b * S_LEN + q0 + r + 8 * hi;
      AO[row * DMODEL + h * DHEAD + 16 * t + ln] = (_Float16)(o[t][r] * inv[r]);
    }
}

// ---------------------------------------------------------------------------
extern "C" void kernel_launch(void* const* d_in, const int* in_sizes, int n_in,
                              void* d_out, int out_size, void* d_ws, size_t ws_size,
                              hipStream_t stream) {
  (void)in_sizes; (void)n_in; (void)out_size; (void)ws_size;
  const float* v  = (const float*)d_in[0];
  const float* k  = (const float*)d_in[1];
  const float* q  = (const float*)d_in[2];
  const float* Wq = (const float*)d_in[3];
  const float* bq = (const float*)d_in[4];
  const float* Wk = (const float*)d_in[5];
  const float* bk = (const float*)d_in[6];
  const float* Wv = (const float*)d_in[7];
  const float* bv = (const float*)d_in[8];
  const float* Wo = (const float*)d_in[9];
  const float* bo = (const float*)d_in[10];

  constexpr int Bn = 2, S = 8192, D = 1024;
  constexpr int M = Bn * S;
  const size_t elems = (size_t)M * D;

  _Float16* Qh = (_Float16*)d_ws;  // 32 MB each, 128 MB total
  _Float16* Kh = Qh + elems;
  _Float16* Vh = Kh + elems;
  _Float16* AO = Vh + elems;

  dim3 blk(256);
  dim3 ggrid(D / 128, M / 128);

  // QKV projections (Q pre-scaled by 1/sqrt(64))
  gemm128<false, true><<<ggrid, blk, 0, stream>>>(q, Wq, bq, Qh, M, D, D, 0.125f);
  gemm128<false, true><<<ggrid, blk, 0, stream>>>(k, Wk, bk, Kh, M, D, D, 1.0f);
  gemm128<false, true><<<ggrid, blk, 0, stream>>>(v, Wv, bv, Vh, M, D, D, 1.0f);

  // attention
  flash_attn<<<dim3(S / 128, 16, Bn), blk, 0, stream>>>(Qh, Kh, Vh, AO);

  // output projection -> f32 d_out
  gemm128<true, false><<<ggrid, blk, 0, stream>>>(AO, Wo, bo, d_out, M, D, D, 1.0f);
}
